// DCANBinaryClassifier_14121852469831
// MI455X (gfx1250) — matrix operand order repaired and checked by reading, not compiled, not run on hardware
//
#include <hip/hip_runtime.h>
#include <hip/hip_bf16.h>

// ---------------------------------------------------------------------------
// DCAN binary classifier pipeline for MI455X (gfx1250, wave32, WMMA + TDM).
// Heavy GEMMs: v_wmma_f32_16x16x32_f16, A-panel staged to LDS by the Tensor
// Data Mover (double-buffered, s_wait_tensorcnt), B = L2-resident
// pre-transposed f16 weights loaded directly. Small ops are VALU kernels.
// ---------------------------------------------------------------------------

typedef __attribute__((ext_vector_type(16))) _Float16 v16h;
typedef __attribute__((ext_vector_type(8)))  _Float16 v8h;
typedef __attribute__((ext_vector_type(8)))  float    v8f;
typedef __attribute__((ext_vector_type(4)))  float    v4f;
typedef unsigned int u32x4 __attribute__((ext_vector_type(4)));
typedef int          i32x4 __attribute__((ext_vector_type(4)));
typedef int          i32x8 __attribute__((ext_vector_type(8)));

union V16 { v16h v; v8h h[2]; };

#if defined(__has_builtin)
#if __has_builtin(__builtin_amdgcn_tensor_load_to_lds)
#define HAVE_TDM 1
#endif
#endif

#ifdef HAVE_TDM
// 2D TDM load: tile [128 rows x 32 cols] of dsz-byte elements, row stride
// rowStrideElems, global tile start gsrc -> LDS byte offset lds_off.
// D# packing per CDNA5 ISA 8.3/8.4 (count=1, type=2 "image", 2D groups only).
__device__ __forceinline__ void tdm_load_tile(const void* gsrc, unsigned lds_off,
                                              int rowStrideElems, int dsz_code) {
  unsigned long long ga = (unsigned long long)(uintptr_t)gsrc;
  u32x4 g0;
  g0.x = 1u;                                            // count=1 (valid D#)
  g0.y = lds_off;                                       // lds_addr
  g0.z = (unsigned)ga;                                  // global_addr[31:0]
  g0.w = (unsigned)((ga >> 32) & 0x01ffffffull)         // global_addr[56:32]
         | (2u << 30);                                  // type=2
  i32x8 g1;
  g1.s0 = dsz_code << 16;                               // data_size, wg_mask=0
  g1.s1 = (32 & 0xffff) << 16;                          // tensor_dim0 = 32
  g1.s2 = (128 & 0xffff) << 16;                         // tensor_dim1 = 128
  g1.s3 = (32 & 0xffff) << 16;                          // tile_dim0 = 32
  g1.s4 = (128 & 0xffff);                               // tile_dim1 = 128
  g1.s5 = rowStrideElems;                               // dim0_stride[31:0]
  g1.s6 = 0;
  g1.s7 = 0;
  i32x4 gz = (i32x4)0;
#if defined(__clang_major__) && __clang_major__ >= 23
  __builtin_amdgcn_tensor_load_to_lds(g0, g1, gz, gz, (i32x8)0, 0);
#else
  __builtin_amdgcn_tensor_load_to_lds(g0, g1, gz, gz, 0);
#endif
}

__device__ __forceinline__ void tdm_wait(int pending) {
#if __has_builtin(__builtin_amdgcn_s_wait_tensorcnt)
  if (pending) __builtin_amdgcn_s_wait_tensorcnt((short)1);
  else         __builtin_amdgcn_s_wait_tensorcnt((short)0);
#else
  if (pending) asm volatile("s_wait_tensorcnt 0x1" ::: "memory");
  else         asm volatile("s_wait_tensorcnt 0x0" ::: "memory");
#endif
}
#endif  // HAVE_TDM

// ---------------------------------------------------------------------------
// Weight prep: f32 [K,N] row-major  ->  f16 [N,K] row-major (transposed)
// ---------------------------------------------------------------------------
__global__ void wt_prep_kernel(const float* __restrict__ W, _Float16* __restrict__ Wt,
                               int K, int N) {
  int idx = blockIdx.x * 256 + threadIdx.x;
  if (idx < K * N) {
    int n = idx / K, k = idx - n * K;
    Wt[idx] = (_Float16)W[(size_t)k * N + n];
  }
}

// ---------------------------------------------------------------------------
// WMMA GEMM: out[T,Ncols] = relu(A[T,Kd] @ W + bias), A f32 or f16,
// W given transposed f16 [Ncols,Kd]. mode 0: store f16. mode 1: masked
// column-sum into psum[B,256] (attention-pool P), no store.
// Block: 256 thr (8 waves), tile 128x256; wave tile 32x128 (2x8 v8f acc).
// A panel (128x32) staged to LDS by TDM, double-buffered.
// ---------------------------------------------------------------------------
__global__ __launch_bounds__(256)
void gemm_relu_wmma(const void* __restrict__ Ain, const _Float16* __restrict__ Wt,
                    const float* __restrict__ bias, _Float16* __restrict__ out,
                    float* __restrict__ psum, const unsigned char* __restrict__ mask,
                    int T, int Kd, int Ncols, int Ltok, int af32, int mode) {
  __shared__ _Float16 rep[8][16][16];       // per-wave repack tile (4 KB)
  __shared__ char As_raw[2][128 * 32 * 4];  // A panel double buffer (32 KB)

  const int tid   = threadIdx.x;
  const int wv    = tid >> 5;
  const int ln    = tid & 31;
  const int lhalf = ln >> 4;                // 0: lanes 0-15, 1: lanes 16-31
  const int l16   = ln & 15;
  const int mrow0 = blockIdx.x * 128;
  const int mbase = (wv & 3) * 32;          // wave's rows within block tile
  const int nbase = blockIdx.y * 256 + (wv >> 2) * 128;

  v8f acc[2][8] = {};

  const int kA = lhalf * 8;                 // A fragment K offset within 32
  const int kB = lhalf * 8;                 // B fragment K offset within 32
  const int esz = af32 ? 4 : 2;
  const int nk  = Kd / 32;
  const char* Abase = (const char*)Ain;

#ifdef HAVE_TDM
  const unsigned lds0 = (unsigned)(uintptr_t)&As_raw[0][0];
  const int bufBytes = 128 * 32 * esz;
  if (wv == 0)
    tdm_load_tile(Abase + (size_t)mrow0 * Kd * esz, lds0, Kd, af32 ? 2 : 1);
#endif

  for (int i = 0; i < nk; ++i) {
    const int k0 = i * 32;
    V16 a[2];

#ifdef HAVE_TDM
    if (wv == 0) {
      if (i + 1 < nk) {
        tdm_load_tile(Abase + ((size_t)mrow0 * Kd + (size_t)(k0 + 32)) * esz,
                      lds0 + (unsigned)(((i + 1) & 1) * bufBytes), Kd, af32 ? 2 : 1);
        tdm_wait(1);
      } else {
        tdm_wait(0);
      }
    }
    __syncthreads();                        // A panel buf[i&1] ready
    {
      const char* Ab = &As_raw[0][0] + (i & 1) * bufBytes;
#pragma unroll
      for (int mt = 0; mt < 2; ++mt) {
        const int r = mbase + mt * 16 + l16;
        if (af32) {
          const float* ap = (const float*)Ab + r * 32 + kA;
          v4f x0 = *(const v4f*)(ap);
          v4f x1 = *(const v4f*)(ap + 4);
          v4f x2 = *(const v4f*)(ap + 16);
          v4f x3 = *(const v4f*)(ap + 20);
#pragma unroll
          for (int j = 0; j < 4; ++j) {
            a[mt].v[j]      = (_Float16)x0[j];
            a[mt].v[4 + j]  = (_Float16)x1[j];
            a[mt].v[8 + j]  = (_Float16)x2[j];
            a[mt].v[12 + j] = (_Float16)x3[j];
          }
        } else {
          const _Float16* ap = (const _Float16*)Ab + r * 32 + kA;
          a[mt].h[0] = *(const v8h*)(ap);
          a[mt].h[1] = *(const v8h*)(ap + 16);
        }
      }
    }
#else
    // Fallback: direct (strided) global loads of the A fragments.
#pragma unroll
    for (int mt = 0; mt < 2; ++mt) {
      const int r = mrow0 + mbase + mt * 16 + l16;
      if (af32) {
        const float* ap = (const float*)Abase + (size_t)r * Kd + k0 + kA;
        v4f x0 = *(const v4f*)(ap);
        v4f x1 = *(const v4f*)(ap + 4);
        v4f x2 = *(const v4f*)(ap + 16);
        v4f x3 = *(const v4f*)(ap + 20);
#pragma unroll
        for (int j = 0; j < 4; ++j) {
          a[mt].v[j]      = (_Float16)x0[j];
          a[mt].v[4 + j]  = (_Float16)x1[j];
          a[mt].v[8 + j]  = (_Float16)x2[j];
          a[mt].v[12 + j] = (_Float16)x3[j];
        }
      } else {
        const _Float16* ap = (const _Float16*)Abase + (size_t)r * Kd + k0 + kA;
        a[mt].h[0] = *(const v8h*)(ap);
        a[mt].h[1] = *(const v8h*)(ap + 16);
      }
    }
#endif

#pragma unroll
    for (int nt = 0; nt < 8; ++nt) {
      const _Float16* bp = Wt + (size_t)(nbase + nt * 16 + l16) * Kd + k0 + kB;
      V16 b;
      b.h[0] = *(const v8h*)(bp);
      b.h[1] = *(const v8h*)(bp + 16);
      acc[0][nt] = __builtin_amdgcn_wmma_f32_16x16x32_f16(
          false, a[0].v, false, b.v, (short)0, acc[0][nt], false, false);
      acc[1][nt] = __builtin_amdgcn_wmma_f32_16x16x32_f16(
          false, a[1].v, false, b.v, (short)0, acc[1][nt], false, false);
    }

#ifdef HAVE_TDM
    __syncthreads();                        // all waves done with buf[i&1]
#endif
  }

  // Epilogue: bias + relu, then masked column-sum (mode 1) or LDS repack ->
  // coalesced b128 f16 stores (mode 0).
#pragma unroll 1
  for (int mt = 0; mt < 2; ++mt) {
    const int mrow = mrow0 + mbase + mt * 16;
#pragma unroll 1
    for (int nt = 0; nt < 8; ++nt) {
      const int ncol = nbase + nt * 16;
      const float bsv = bias[ncol + l16];
#pragma unroll
      for (int r = 0; r < 8; ++r) {
        const int rowLocal = r + lhalf * 8;
        float v = acc[mt][nt][r] + bsv;
        v = fmaxf(v, 0.0f);
        if (mode == 1) {
          const int grow = mrow + rowLocal;
          if (mask[grow]) atomicAdd(&psum[(grow / Ltok) * 256 + ncol + l16], v);
        } else {
          rep[wv][rowLocal][l16] = (_Float16)v;
        }
      }
      if (mode == 0) {
        asm volatile("s_wait_dscnt 0" ::: "memory");
        const int row = ln >> 1;
        const int co  = (ln & 1) * 8;
        v8h pk = *(const v8h*)&rep[wv][row][co];
        *(v8h*)(out + (size_t)(mrow + row) * Ncols + ncol + co) = pk;
        asm volatile("s_wait_dscnt 0" ::: "memory");
      }
    }
  }
}

// ---------------------------------------------------------------------------
// LayerNorm with attr injection, in place on f16 [T,256]:
//   y = LN(x + attr[row]*Wa + ba) * g + be.  One block (256 thr) per row.
// ---------------------------------------------------------------------------
__global__ __launch_bounds__(256)
void ln_attr_kernel(_Float16* __restrict__ x, const float* __restrict__ attr,
                    const float* __restrict__ Wa, const float* __restrict__ ba,
                    const float* __restrict__ g, const float* __restrict__ be) {
  __shared__ float red[8];
  const int r = blockIdx.x, t = threadIdx.x;
  float v = (float)x[(size_t)r * 256 + t] + attr[r] * Wa[t] + ba[t];

  float s = v;
  for (int o = 16; o > 0; o >>= 1) s += __shfl_xor(s, o, 32);
  if ((t & 31) == 0) red[t >> 5] = s;
  __syncthreads();
  float sum = 0.f;
  for (int i = 0; i < 8; ++i) sum += red[i];
  const float mean = sum * (1.0f / 256.0f);
  __syncthreads();

  const float d = v - mean;
  float sq = d * d;
  for (int o = 16; o > 0; o >>= 1) sq += __shfl_xor(sq, o, 32);
  if ((t & 31) == 0) red[t >> 5] = sq;
  __syncthreads();
  float var = 0.f;
  for (int i = 0; i < 8; ++i) var += red[i];
  var *= (1.0f / 256.0f);

  x[(size_t)r * 256 + t] = (_Float16)(d * rsqrtf(var + 1e-5f) * g[t] + be[t]);
}

// ---------------------------------------------------------------------------
// Finalize masked mean: pooled[b,c] = psum[b,c] / max(count(mask_b), 1)
// ---------------------------------------------------------------------------
__global__ __launch_bounds__(256)
void pool_kernel(const float* __restrict__ psum, const unsigned char* __restrict__ mask,
                 int Ltok, float* __restrict__ pooled) {
  __shared__ float red[8];
  const int b = blockIdx.x, t = threadIdx.x;
  float c = 0.f;
  for (int tok = t; tok < Ltok; tok += 256) c += mask[b * Ltok + tok] ? 1.f : 0.f;
  for (int o = 16; o > 0; o >>= 1) c += __shfl_xor(c, o, 32);
  if ((t & 31) == 0) red[t >> 5] = c;
  __syncthreads();
  float cnt = 0.f;
  for (int i = 0; i < 8; ++i) cnt += red[i];
  pooled[b * 256 + t] = psum[b * 256 + t] / fmaxf(cnt, 1.f);
}

// ---------------------------------------------------------------------------
// Single-query attention per batch row:
//   out[b] = softmax(K[b] q[b] / 16, mask) @ V[b]   (nan_to_num all-masked)
// One block (256 thr) per batch. Ltok <= 512.
// ---------------------------------------------------------------------------
__global__ __launch_bounds__(256)
void attn_kernel(const float* __restrict__ qin, const _Float16* __restrict__ K,
                 const _Float16* __restrict__ V, const unsigned char* __restrict__ mask,
                 int Ltok, float* __restrict__ out) {
  __shared__ float qs[256];
  __shared__ float sc[512];
  __shared__ float red[8];
  const int b = blockIdx.x, t = threadIdx.x;
  qs[t] = qin[b * 256 + t];
  __syncthreads();

  const float NEG = -__builtin_inff();
  float lmax = NEG;
  for (int tok = t; tok < Ltok; tok += 256) {
    const _Float16* kr = K + ((size_t)b * Ltok + tok) * 256;
    float s = 0.f;
    for (int c = 0; c < 256; ++c) s += qs[c] * (float)kr[c];
    s *= 0.0625f;                                   // 1/sqrt(256)
    if (!mask[b * Ltok + tok]) s = NEG;
    sc[tok] = s;
    lmax = fmaxf(lmax, s);
  }
  for (int o = 16; o > 0; o >>= 1) lmax = fmaxf(lmax, __shfl_xor(lmax, o, 32));
  if ((t & 31) == 0) red[t >> 5] = lmax;
  __syncthreads();
  float mx = NEG;
  for (int i = 0; i < 8; ++i) mx = fmaxf(mx, red[i]);
  __syncthreads();

  float lsum = 0.f;
  for (int tok = t; tok < Ltok; tok += 256) {
    float e = (mx == NEG) ? 0.f : __expf(sc[tok] - mx);
    sc[tok] = e;
    lsum += e;
  }
  for (int o = 16; o > 0; o >>= 1) lsum += __shfl_xor(lsum, o, 32);
  if ((t & 31) == 0) red[t >> 5] = lsum;
  __syncthreads();
  float den = 0.f;
  for (int i = 0; i < 8; ++i) den += red[i];
  const float inv = den > 0.f ? 1.f / den : 0.f;    // nan_to_num path

  float acc = 0.f;
  for (int tok = 0; tok < Ltok; ++tok)
    acc += sc[tok] * (float)V[((size_t)b * Ltok + tok) * 256 + t];
  out[b * 256 + t] = acc * inv;
}

// ---------------------------------------------------------------------------
// Cross-block update: q[r] += ctx[r] @ W + b      (W: f32 [256,256])
// ---------------------------------------------------------------------------
__global__ __launch_bounds__(256)
void cb_update_kernel(float* __restrict__ q, const float* __restrict__ ctx,
                      const float* __restrict__ W, const float* __restrict__ bvec) {
  __shared__ float cr[256];
  const int r = blockIdx.x, t = threadIdx.x;
  cr[t] = ctx[r * 256 + t];
  __syncthreads();
  float s = 0.f;
  for (int k = 0; k < 256; ++k) s += cr[k] * W[k * 256 + t];
  q[r * 256 + t] += s + bvec[t];
}

// ---------------------------------------------------------------------------
// fused[r] = concat(img_q[r], txt_q[r], relu(meta[r]@Wm + bm)) as f16
// ---------------------------------------------------------------------------
__global__ __launch_bounds__(256)
void build_fused_kernel(const float* __restrict__ img_q, const float* __restrict__ txt_q,
                        const float* __restrict__ meta, const float* __restrict__ Wm,
                        const float* __restrict__ bm, _Float16* __restrict__ fused) {
  __shared__ float mrow[64];
  const int r = blockIdx.x, t = threadIdx.x;
  if (t < 64) mrow[t] = meta[r * 64 + t];
  __syncthreads();
  float s = 0.f;
  for (int k = 0; k < 64; ++k) s += mrow[k] * Wm[k * 256 + t];
  s = fmaxf(s + bm[t], 0.f);
  fused[(size_t)r * 768 + t]       = (_Float16)img_q[r * 256 + t];
  fused[(size_t)r * 768 + 256 + t] = (_Float16)txt_q[r * 256 + t];
  fused[(size_t)r * 768 + 512 + t] = (_Float16)s;
}

// ---------------------------------------------------------------------------
// Head: out[r] = z[r,:1536] . Whead + bhead
// ---------------------------------------------------------------------------
__global__ __launch_bounds__(256)
void head_kernel(const _Float16* __restrict__ z, const float* __restrict__ Wh,
                 const float* __restrict__ bh, float* __restrict__ outp) {
  __shared__ float red[8];
  const int r = blockIdx.x, t = threadIdx.x;
  float s = 0.f;
  for (int j = 0; j < 6; ++j) {
    const int c = t + j * 256;
    s += (float)z[(size_t)r * 1536 + c] * Wh[c];
  }
  for (int o = 16; o > 0; o >>= 1) s += __shfl_xor(s, o, 32);
  if ((t & 31) == 0) red[t >> 5] = s;
  __syncthreads();
  if (t == 0) {
    float tot = 0.f;
    for (int i = 0; i < 8; ++i) tot += red[i];
    outp[r] = tot + bh[0];
  }
}

// ---------------------------------------------------------------------------
extern "C" void kernel_launch(void* const* d_in, const int* in_sizes, int n_in,
                              void* d_out, int out_size, void* d_ws, size_t ws_size,
                              hipStream_t stream) {
  (void)in_sizes; (void)n_in; (void)out_size; (void)ws_size;

  const float* img_emb  = (const float*)d_in[0];
  const float* img_attr = (const float*)d_in[1];
  const float* txt_emb  = (const float*)d_in[2];
  const float* txt_attr = (const float*)d_in[3];
  const unsigned char* img_mask = (const unsigned char*)d_in[4];
  const unsigned char* txt_mask = (const unsigned char*)d_in[5];
  const float* meta   = (const float*)d_in[6];
  const float* Wi     = (const float*)d_in[7];
  const float* bi     = (const float*)d_in[8];
  const float* Wtp    = (const float*)d_in[9];
  const float* btp    = (const float*)d_in[10];
  const float* Wia    = (const float*)d_in[11];
  const float* bia    = (const float*)d_in[12];
  const float* Wta    = (const float*)d_in[13];
  const float* bta    = (const float*)d_in[14];
  const float* g_img  = (const float*)d_in[15];
  const float* be_img = (const float*)d_in[16];
  const float* g_txt  = (const float*)d_in[17];
  const float* be_txt = (const float*)d_in[18];
  const float* Wqg_img = (const float*)d_in[19];
  const float* bqg_img = (const float*)d_in[20];
  const float* Wqg_txt = (const float*)d_in[21];
  const float* bqg_txt = (const float*)d_in[22];
  const float* Wcb_img = (const float*)d_in[23];
  const float* bcb_img = (const float*)d_in[24];
  const float* Wcb_txt = (const float*)d_in[25];
  const float* bcb_txt = (const float*)d_in[26];
  const float* Wmeta  = (const float*)d_in[27];
  const float* bmeta  = (const float*)d_in[28];
  const float* Wfus   = (const float*)d_in[29];
  const float* bfus   = (const float*)d_in[30];
  const float* Whead  = (const float*)d_in[31];
  const float* bhead  = (const float*)d_in[32];
  float* out = (float*)d_out;

  constexpr int B = 128, N = 196, M = 512, Di = 1024, Dt = 768, D = 256;
  const int Ti = B * N;   // 25088 = 196 * 128
  const int Tt = B * M;   // 65536 = 512 * 128

  // ---- workspace bump allocator (256B aligned) ----
  char* w = (char*)d_ws;
  auto alloc = [&](size_t bytes) -> char* {
    char* p = w;
    w += (bytes + 255) & ~(size_t)255;
    return p;
  };
  _Float16* Wt_i  = (_Float16*)alloc((size_t)D * Di * 2);
  _Float16* Wt_t  = (_Float16*)alloc((size_t)D * Dt * 2);
  _Float16* Wt_qi = (_Float16*)alloc((size_t)3 * D * D * 2);
  _Float16* Wt_qt = (_Float16*)alloc((size_t)3 * D * D * 2);
  _Float16* Wt_f  = (_Float16*)alloc((size_t)1536 * 768 * 2);
  _Float16* img_x = (_Float16*)alloc((size_t)Ti * D * 2);
  _Float16* txt_x = (_Float16*)alloc((size_t)Tt * D * 2);
  _Float16* K_img = (_Float16*)alloc((size_t)Ti * D * 2);
  _Float16* V_img = (_Float16*)alloc((size_t)Ti * D * 2);
  _Float16* K_txt = (_Float16*)alloc((size_t)Tt * D * 2);
  _Float16* V_txt = (_Float16*)alloc((size_t)Tt * D * 2);
  float* psum_img = (float*)alloc((size_t)B * D * 4);
  float* psum_txt = (float*)alloc((size_t)B * D * 4);
  float* pooled_i = (float*)alloc((size_t)B * D * 4);
  float* pooled_t = (float*)alloc((size_t)B * D * 4);
  float* img_q    = (float*)alloc((size_t)B * D * 4);
  float* txt_q    = (float*)alloc((size_t)B * D * 4);
  float* ctx      = (float*)alloc((size_t)B * D * 4);
  _Float16* fused = (_Float16*)alloc((size_t)B * 768 * 2);
  _Float16* zbuf  = (_Float16*)alloc((size_t)B * 1536 * 2);

  const dim3 blk(256);
  auto cdiv = [](int a, int b) { return (a + b - 1) / b; };

  // ---- 1. weight transposition to f16 ----
  wt_prep_kernel<<<cdiv(Di * D, 256), blk, 0, stream>>>(Wi, Wt_i, Di, D);
  wt_prep_kernel<<<cdiv(Dt * D, 256), blk, 0, stream>>>(Wtp, Wt_t, Dt, D);
  for (int wi = 0; wi < 3; ++wi) {
    wt_prep_kernel<<<cdiv(D * D, 256), blk, 0, stream>>>(Wqg_img + wi * D * D, Wt_qi + wi * D * D, D, D);
    wt_prep_kernel<<<cdiv(D * D, 256), blk, 0, stream>>>(Wqg_txt + wi * D * D, Wt_qt + wi * D * D, D, D);
  }
  wt_prep_kernel<<<cdiv(768 * 1536, 256), blk, 0, stream>>>(Wfus, Wt_f, 768, 1536);

  hipMemsetAsync(psum_img, 0, (size_t)B * D * 4, stream);
  hipMemsetAsync(psum_txt, 0, (size_t)B * D * 4, stream);

  // ---- 2. input projections (f32 A path, TDM-staged) ----
  gemm_relu_wmma<<<dim3(Ti / 128, 1), blk, 0, stream>>>(
      img_emb, Wt_i, bi, img_x, nullptr, nullptr, Ti, Di, D, N, 1, 0);
  gemm_relu_wmma<<<dim3(Tt / 128, 1), blk, 0, stream>>>(
      txt_emb, Wt_t, btp, txt_x, nullptr, nullptr, Tt, Dt, D, M, 1, 0);

  // ---- 3. attr + LayerNorm (in place) ----
  ln_attr_kernel<<<Ti, blk, 0, stream>>>(img_x, img_attr, Wia, bia, g_img, be_img);
  ln_attr_kernel<<<Tt, blk, 0, stream>>>(txt_x, txt_attr, Wta, bta, g_txt, be_txt);

  // ---- 4. qgen GEMMs: K, V stored; P masked-summed into psum ----
  gemm_relu_wmma<<<dim3(Ti / 128, 1), blk, 0, stream>>>(
      img_x, Wt_qi + 0 * D * D, bqg_img + 0 * D, K_img, nullptr, nullptr, Ti, D, D, N, 0, 0);
  gemm_relu_wmma<<<dim3(Ti / 128, 1), blk, 0, stream>>>(
      img_x, Wt_qi + 1 * D * D, bqg_img + 1 * D, V_img, nullptr, nullptr, Ti, D, D, N, 0, 0);
  gemm_relu_wmma<<<dim3(Ti / 128, 1), blk, 0, stream>>>(
      img_x, Wt_qi + 2 * D * D, bqg_img + 2 * D, nullptr, psum_img, img_mask, Ti, D, D, N, 0, 1);
  gemm_relu_wmma<<<dim3(Tt / 128, 1), blk, 0, stream>>>(
      txt_x, Wt_qt + 0 * D * D, bqg_txt + 0 * D, K_txt, nullptr, nullptr, Tt, D, D, M, 0, 0);
  gemm_relu_wmma<<<dim3(Tt / 128, 1), blk, 0, stream>>>(
      txt_x, Wt_qt + 1 * D * D, bqg_txt + 1 * D, V_txt, nullptr, nullptr, Tt, D, D, M, 0, 0);
  gemm_relu_wmma<<<dim3(Tt / 128, 1), blk, 0, stream>>>(
      txt_x, Wt_qt + 2 * D * D, bqg_txt + 2 * D, nullptr, psum_txt, txt_mask, Tt, D, D, M, 0, 1);

  // ---- 5. masked mean -> pooled queries -> qgen attention ----
  pool_kernel<<<B, blk, 0, stream>>>(psum_img, img_mask, N, pooled_i);
  pool_kernel<<<B, blk, 0, stream>>>(psum_txt, txt_mask, M, pooled_t);
  attn_kernel<<<B, blk, 0, stream>>>(pooled_i, K_img, V_img, img_mask, N, img_q);
  attn_kernel<<<B, blk, 0, stream>>>(pooled_t, K_txt, V_txt, txt_mask, M, txt_q);

  // ---- 6. cross-modal attention blocks (L=2, sequential) ----
  for (int l = 0; l < 2; ++l) {
    attn_kernel<<<B, blk, 0, stream>>>(txt_q, K_img, V_img, img_mask, N, ctx);
    cb_update_kernel<<<B, blk, 0, stream>>>(img_q, ctx, Wcb_img + l * D * D, bcb_img + l * D);
    attn_kernel<<<B, blk, 0, stream>>>(img_q, K_txt, V_txt, txt_mask, M, ctx);
    cb_update_kernel<<<B, blk, 0, stream>>>(txt_q, ctx, Wcb_txt + l * D * D, bcb_txt + l * D);
  }

  // ---- 7. meta + fusion + head ----
  build_fused_kernel<<<B, blk, 0, stream>>>(img_q, txt_q, meta, Wmeta, bmeta, fused);
  gemm_relu_wmma<<<dim3(B / 128, 1536 / 256), blk, 0, stream>>>(
      fused, Wt_f, bfus, zbuf, nullptr, nullptr, B, 768, 1536, B, 0, 0);
  head_kernel<<<B, blk, 0, stream>>>(zbuf, Whead, bhead, out);
}
